// GraphAttentionLayer_30279519437044
// MI455X (gfx1250) — matrix operand order repaired and checked by reading, not compiled, not run on hardware
//
#include <hip/hip_runtime.h>
#include <stdint.h>

// ---------------------------------------------------------------------------
// GAT layer, fused for MI455X (gfx1250, wave32, WMMA f16)
//   h   = inp @ W                          (WMMA, f16 in / f32 acc)
//   s1  = h @ a1 ; s2 = h @ a2
//   e   = lrelu(s1_i + s2_j); masked softmax over j; out = elu(att @ h)
// Single streaming pass over adj (256 MB) fused with the att@h WMMA GEMM.
// Double-buffered async-DMA h-tiles, one barrier per 64-column step.
// ---------------------------------------------------------------------------

#define NN    8192
#define INF_  256
#define OUTF  256
#define ALPHA 0.2f

typedef __attribute__((ext_vector_type(16))) _Float16 v16h;
typedef __attribute__((ext_vector_type(8)))  float    v8f;
typedef __attribute__((ext_vector_type(4)))  int      v4i;

#if defined(__has_builtin)
#if __has_builtin(__builtin_amdgcn_global_load_async_to_lds_b128) && \
    __has_builtin(__builtin_amdgcn_s_wait_asynccnt)
#define USE_ASYNC_LDS 1
#endif
#endif

#ifdef USE_ASYNC_LDS
typedef __attribute__((address_space(1))) v4i gv4i;   // global-AS int4
typedef __attribute__((address_space(3))) v4i lv4i;   // LDS-AS int4
#endif

static __device__ __forceinline__ uint16_t f2h_bits(float f) {
  union { _Float16 h; uint16_t u; } v; v.h = (_Float16)f;   // v_cvt_f16_f32
  return v.u;
}
static __device__ __forceinline__ float h2f(uint16_t b) {
  union { uint16_t u; _Float16 h; } v; v.u = b;
  return (float)v.h;
}

// ---------------------------------------------------------------------------
// Kernel 1: h = inp @ W, stored transposed as f16:  h_t[f][i]  (256 x 8192)
// Block = 128 threads (4 waves). Wave w: rows i0 = bx*64 + 16w, cols by*64..+63.
// ---------------------------------------------------------------------------
__global__ void __launch_bounds__(128)
gat_h_gemm(const float* __restrict__ inp, const float* __restrict__ W,
           uint16_t* __restrict__ h_t) {
  const int lane = threadIdx.x & 31;
  const int wave = threadIdx.x >> 5;
  const int i0   = blockIdx.x * 64 + wave * 16;
  const int c0   = blockIdx.y * 64;
  const int m    = lane & 15;
  const int half = lane >> 4;
  const int hb   = half * 8;     // A-fragment K chunk base (ISA 16-bit A layout)
  const int kb   = half * 16;    // unused; B fragment uses hb, same k order as A

  v8f acc[4] = {};

  const float* arow = inp + (size_t)(i0 + m) * INF_;

  for (int k = 0; k < INF_; k += 32) {
    // ---- A fragment: rows of inp, f32 -> f16, per ISA 16-bit A layout ----
    union { uint16_t u[16]; v16h v; } af;
    float4 a0 = *(const float4*)(arow + k + hb);
    float4 a1 = *(const float4*)(arow + k + hb + 4);
    float4 a2 = *(const float4*)(arow + k + hb + 16);
    float4 a3 = *(const float4*)(arow + k + hb + 20);
    af.u[0]=f2h_bits(a0.x); af.u[1]=f2h_bits(a0.y); af.u[2]=f2h_bits(a0.z); af.u[3]=f2h_bits(a0.w);
    af.u[4]=f2h_bits(a1.x); af.u[5]=f2h_bits(a1.y); af.u[6]=f2h_bits(a1.z); af.u[7]=f2h_bits(a1.w);
    af.u[8]=f2h_bits(a2.x); af.u[9]=f2h_bits(a2.y); af.u[10]=f2h_bits(a2.z); af.u[11]=f2h_bits(a2.w);
    af.u[12]=f2h_bits(a3.x); af.u[13]=f2h_bits(a3.y); af.u[14]=f2h_bits(a3.z); af.u[15]=f2h_bits(a3.w);

    #pragma unroll
    for (int t = 0; t < 4; ++t) {
      const int c = c0 + t * 16 + m;
      union { uint16_t u[16]; v16h v; } bf;
      #pragma unroll
      for (int e = 0; e < 16; ++e)
        bf.u[e] = f2h_bits(W[(size_t)(k + hb + e + (e >= 8 ? 8 : 0)) * OUTF + c]);  // W is L2-resident
      acc[t] = __builtin_amdgcn_wmma_f32_16x16x32_f16(
          false, af.v, false, bf.v, (short)0, acc[t], false, false);
    }
  }

  // ---- store transposed f16 (C/D layout: row = r + 8*half, col = t*16+m) ----
  #pragma unroll
  for (int t = 0; t < 4; ++t) {
    const int c = c0 + t * 16 + m;
    #pragma unroll
    for (int r = 0; r < 8; ++r) {
      const int i = i0 + r + 8 * half;
      h_t[(size_t)c * NN + i] = f2h_bits(acc[t][r]);
    }
  }
}

// ---------------------------------------------------------------------------
// Kernel 2a: init global-max cell (ws is poisoned by harness)
// ---------------------------------------------------------------------------
__global__ void gat_init(int* s2max_enc) {
  if (threadIdx.x == 0) *s2max_enc = (int)0x80000000;
}

// ---------------------------------------------------------------------------
// Kernel 2b: s1 = h@a1, s2 = h@a2; global max of s2 via monotonic-int atomicMax.
// ---------------------------------------------------------------------------
__global__ void __launch_bounds__(256)
gat_scores(const uint16_t* __restrict__ h_t, const float* __restrict__ a,
           float* __restrict__ s1, float* __restrict__ s2,
           int* __restrict__ s2max_enc) {
  const int i = blockIdx.x * blockDim.x + threadIdx.x;
  float acc1 = 0.f, acc2 = 0.f;
  for (int f = 0; f < OUTF; ++f) {
    const float hv = h2f(h_t[(size_t)f * NN + i]);
    acc1 = fmaf(hv, a[f],        acc1);
    acc2 = fmaf(hv, a[OUTF + f], acc2);
  }
  s1[i] = acc1;
  s2[i] = acc2;
  int enc = __float_as_int(acc2);
  enc = (enc >= 0) ? enc : (enc ^ 0x7fffffff);   // monotonic float->int encoding
  atomicMax(s2max_enc, enc);
}

// ---------------------------------------------------------------------------
// Kernel 3: fused masked-softmax + att@h + ELU.
// Block = 256 threads (8 waves): rg = wave>>2 picks 16-row group, fg = wave&3
// picks 64 output features. 64 attention columns per step. Per step:
//   produce p(it)  ->  s_wait_asynccnt 0  ->  barrier  ->
//   issue async DMA of tile it+1 into the other buffer  ->  8 WMMAs on tile it
// h-tile and p-fragments double-buffered: ONE barrier per step, DMA overlaps
// a full step of WMMA + adj/exp work. adj streamed exactly once.
// ---------------------------------------------------------------------------
__global__ void __launch_bounds__(256)
gat_attn(const int* __restrict__ adj, const uint16_t* __restrict__ h_t,
         const float* __restrict__ s1, const float* __restrict__ s2,
         const int* __restrict__ s2max_enc, float* __restrict__ out) {
  __shared__ __align__(16) uint16_t sh_h[2][OUTF * 64];      // 2 x 32 KB f16 tiles
  __shared__ __align__(16) uint16_t sh_p[2][2][2][16 * 32];  // [buf][rg][s] A-frags
  __shared__ float sh_lp[2][2][16];                          // partial denominators

  const int lane = threadIdx.x & 31;
  const int wave = threadIdx.x >> 5;
  const int rg   = wave >> 2;             // row group 0/1
  const int fg   = wave & 3;              // feature group 0..3 (fg 0/1 also produce p)
  const int i0   = blockIdx.x * 32 + rg * 16;
  const int m    = lane & 15;
  const int half = lane >> 4;
  const int hb   = half * 8;              // A-layout K chunk base
  const int kb   = half * 16;             // unused; B frags use hb (same k order as A)
  const int fb   = fg * 64;

  // upper bound of every e_ij for this row (lrelu is monotonic)
  const int enc  = *s2max_enc;
  const float s2max = __int_as_float(enc >= 0 ? enc : (enc ^ 0x7fffffff));
  const int   irow  = i0 + m;
  const float s1v   = s1[irow];
  float mx = s1v + s2max;
  mx = (mx >= 0.f) ? mx : ALPHA * mx;

  const int* adjrow = adj + (size_t)irow * NN;

  v8f acc[4] = {};
  float psum = 0.f;

  // stage h_t[f][j..j+63] (128 B per thread, f = threadIdx.x) into buffer `buf`
  auto issue_tile = [&](int j, int buf) {
#ifdef USE_ASYNC_LDS
    gv4i* src = (gv4i*)(h_t + (size_t)threadIdx.x * NN + j);
    lv4i* dst = (lv4i*)(sh_h[buf] + threadIdx.x * 64);
    #pragma unroll
    for (int q = 0; q < 8; ++q)
      __builtin_amdgcn_global_load_async_to_lds_b128(src + q, dst + q, 0, 0);
#else
    const uint4* src = (const uint4*)(h_t + (size_t)threadIdx.x * NN + j);
    uint4* dst = (uint4*)(sh_h[buf] + threadIdx.x * 64);
    #pragma unroll
    for (int q = 0; q < 8; ++q) dst[q] = src[q];
#endif
  };

  issue_tile(0, 0);   // prologue: DMA tile 0

  const int NITER = NN / 64;
  for (int it = 0; it < NITER; ++it) {
    const int cur = it & 1;
    const int j   = it * 64;

    // ---- fg==0/1 waves: build p A-fragment for K chunk fg (mask + exp) ----
    if (fg < 2) {
      const int jb = j + fg * 32;
      int   av[16];
      float sv[16];
      const int4*   a4 = (const int4*)(adjrow + jb + hb);
      const float4* s4 = (const float4*)(s2 + jb + hb);
      *(int4*)&av[0]  = a4[0]; *(int4*)&av[4]  = a4[1];   // k = hb .. hb+7
      *(int4*)&av[8]  = a4[4]; *(int4*)&av[12] = a4[5];   // k = hb+16 .. hb+23
      *(float4*)&sv[0]  = s4[0]; *(float4*)&sv[4]  = s4[1];
      *(float4*)&sv[8]  = s4[4]; *(float4*)&sv[12] = s4[5];

      union { uint16_t u[16]; uint4 q[2]; } pf;
      #pragma unroll
      for (int e = 0; e < 16; ++e) {
        float ev = s1v + sv[e];
        ev = (ev >= 0.f) ? ev : ALPHA * ev;
        const float p = (av[e] > 0) ? __expf(ev - mx) : 0.0f;
        psum += p;
        pf.u[e] = f2h_bits(p);
      }
      uint4* pd = (uint4*)(sh_p[cur][rg][fg] + lane * 16);
      pd[0] = pf.q[0]; pd[1] = pf.q[1];

      __builtin_prefetch(adjrow + j + 2048, 0, 1);   // stream adj ahead
    }

#ifdef USE_ASYNC_LDS
    __builtin_amdgcn_s_wait_asynccnt(0);   // tile `it` fully landed in LDS
#endif
    __syncthreads();                       // ONE barrier per step

    // ---- DMA next tile into the other buffer (overlaps the WMMAs below) ----
    if (it + 1 < NITER) issue_tile(j + 64, cur ^ 1);

    // ---- A fragments from LDS, B fragments from staged tile, 8 WMMAs ----
    union { uint4 q[2]; v16h v; } af[2];
    #pragma unroll
    for (int s = 0; s < 2; ++s) {
      const uint4* ps = (const uint4*)(sh_p[cur][rg][s] + lane * 16);
      af[s].q[0] = ps[0]; af[s].q[1] = ps[1];
    }
    #pragma unroll
    for (int t = 0; t < 4; ++t) {
      const int c = fb + t * 16 + m;
      #pragma unroll
      for (int s = 0; s < 2; ++s) {
        union { uint4 q[2]; v16h v; } bfm;
        const uint4* bs = (const uint4*)(sh_h[cur] + c * 64 + s * 32 + hb);
        bfm.q[0] = bs[0]; bfm.q[1] = bs[2];
        acc[t] = __builtin_amdgcn_wmma_f32_16x16x32_f16(
            false, af[s].v, false, bfm.v, (short)0, acc[t], false, false);
      }
    }
  }

  // ---- softmax denominators: combine half-lane partials, then both chunks ----
  if (fg < 2) {
    const float other = __shfl_xor(psum, 16, 32);
    const float l = psum + other;
    if (lane < 16) sh_lp[rg][fg][lane] = l;
  }
  __syncthreads();

  // ---- epilogue: divide, ELU, store ----
  #pragma unroll
  for (int t = 0; t < 4; ++t) {
    const int c = fb + t * 16 + m;
    #pragma unroll
    for (int r = 0; r < 8; ++r) {
      const int row = r + 8 * half;
      const float l = sh_lp[rg][0][row] + sh_lp[rg][1][row];
      const float d = acc[t][r] / l;
      const float o = (d > 0.f) ? d : (__expf(d) - 1.0f);
      out[(size_t)(i0 + row) * OUTF + c] = o;
    }
  }
}

// ---------------------------------------------------------------------------
extern "C" void kernel_launch(void* const* d_in, const int* in_sizes, int n_in,
                              void* d_out, int out_size, void* d_ws, size_t ws_size,
                              hipStream_t stream) {
  const float* inp = (const float*)d_in[0];   // (8192, 256) f32
  const int*   adj = (const int*)  d_in[1];   // (8192, 8192) i32
  const float* W   = (const float*)d_in[2];   // (256, 256) f32
  const float* a   = (const float*)d_in[3];   // (512, 1) f32
  float* out = (float*)d_out;                 // (8192, 256) f32

  char* ws = (char*)d_ws;
  uint16_t* h_t   = (uint16_t*)ws;                          // 4 MB f16, transposed
  float*    s1    = (float*)(ws + (size_t)4 * 1024 * 1024); // 32 KB
  float*    s2    = s1 + NN;                                // 32 KB
  int*      s2max = (int*)(s2 + NN);                        // 4 B

  gat_h_gemm<<<dim3(NN / 64, OUTF / 64), 128, 0, stream>>>(inp, W, h_t);
  gat_init  <<<1, 32, 0, stream>>>(s2max);
  gat_scores<<<NN / 256, 256, 0, stream>>>(h_t, a, s1, s2, s2max);
  gat_attn  <<<NN / 32, 256, 0, stream>>>(adj, h_t, s1, s2, s2max, out);
}
